// TransformerBlock_39402029974039
// MI455X (gfx1250) — compile-verified, matrix operand order checked by reading, not run on hardware
//
#include <hip/hip_runtime.h>
#include <hip/hip_bf16.h>

// ---------------------------------------------------------------------------
// Types
// ---------------------------------------------------------------------------
typedef __bf16 bf16;
typedef __attribute__((ext_vector_type(16))) __bf16 v16bf;
typedef __attribute__((ext_vector_type(8)))  __bf16 bf16x8;
typedef __attribute__((ext_vector_type(8)))  float  v8f;

__device__ __forceinline__ bf16 f2bf(float f) {
  unsigned int u = __builtin_bit_cast(unsigned int, f);
  unsigned int r = u + 0x7FFFu + ((u >> 16) & 1u);          // round-to-nearest-even
  unsigned short s = (unsigned short)(r >> 16);
  return __builtin_bit_cast(bf16, s);
}

// CDNA5 async copy: global -> LDS, 16B per lane, tracked by ASYNCcnt.
// lds_addr: wave-relative LDS byte address (low 32 bits of generic pointer).
__device__ __forceinline__ void async_load_b128(unsigned lds_addr,
                                                unsigned long long gaddr) {
  asm volatile("global_load_async_to_lds_b128 %0, %1, off"
               :: "v"(lds_addr), "v"(gaddr) : "memory");
}
__device__ __forceinline__ void wait_async0() {
  asm volatile("s_wait_asynccnt 0" ::: "memory");
}

// ---------------------------------------------------------------------------
// WMMA GEMM:  C[M x Ncol] = A[M x K] (bf16, row-major) * Bt[Ncol x K]^T (bf16)
// Block tile 128x128, K-step 32, 8 waves (256 thr), wave tile 32x64.
// Global->LDS via async-to-LDS (ASYNCcnt), double-buffered; 8 v_wmma / K-step.
// Epilogue: C = act(acc*scale + bias), written f32 or bf16.
// grid = (Ncol/128, M/128, batch)
// ---------------------------------------------------------------------------
template<bool OUT_BF16, bool RELU, bool HAS_BIAS>
__global__ __launch_bounds__(256)
void gemm_wmma(const bf16* __restrict__ A, const bf16* __restrict__ Bt,
               const float* __restrict__ bias,
               float* __restrict__ Cf, bf16* __restrict__ Cb,
               int M, int Ncol, int K, int lda, int ldb, int ldc,
               long long bsA, long long bsB, long long bsC, float scale)
{
  __shared__ bf16 shA[2][128 * 40];   // stride 40 (80B) -> conflict-free row reads
  __shared__ bf16 shB[2][128 * 40];

  A  += (long long)blockIdx.z * bsA;
  Bt += (long long)blockIdx.z * bsB;
  if (OUT_BF16) Cb += (long long)blockIdx.z * bsC;
  else          Cf += (long long)blockIdx.z * bsC;

  const int row0 = blockIdx.y * 128;
  const int col0 = blockIdx.x * 128;
  const int tid  = threadIdx.x;
  const int lane = tid & 31;
  const int wave = tid >> 5;
  const int wm   = (wave & 3) * 32;   // wave row offset inside block tile
  const int wn   = (wave >> 2) * 64;  // wave col offset inside block tile
  const int half = lane >> 4;         // 0: lanes 0-15, 1: lanes 16-31
  const int l16  = lane & 15;
  const int kb   = half * 8;          // K sub-offset per ISA 16-bit A/B layout

  v8f acc[2][4];
#pragma unroll
  for (int i = 0; i < 2; ++i)
#pragma unroll
    for (int j = 0; j < 4; ++j) acc[i][j] = (v8f)(0.0f);

  // Each thread moves 2 chunks of 8 bf16 per tile (A and B): 4 async ops.
  auto issue_tiles = [&](int bufi, int k0) {
#pragma unroll
    for (int t = 0; t < 2; ++t) {
      int c  = tid + t * 256;   // 512 chunks of 8 bf16 per tile
      int r  = c >> 2;          // 0..127
      int kc = (c & 3) << 3;    // 0,8,16,24
      async_load_b128((unsigned)(size_t)&shA[bufi][r * 40 + kc],
                      (unsigned long long)(size_t)(A  + (size_t)(row0 + r) * lda + (k0 + kc)));
      async_load_b128((unsigned)(size_t)&shB[bufi][r * 40 + kc],
                      (unsigned long long)(size_t)(Bt + (size_t)(col0 + r) * ldb + (k0 + kc)));
    }
  };

  auto compute = [&](int bufi) {
    v16bf af[2], bfr[4];
#pragma unroll
    for (int i = 0; i < 2; ++i) {
      const bf16* p = &shA[bufi][(wm + i * 16 + l16) * 40 + kb];
      bf16x8 lo = *(const bf16x8*)p;
      bf16x8 hi = *(const bf16x8*)(p + 16);
      af[i] = __builtin_shufflevector(lo, hi, 0,1,2,3,4,5,6,7,8,9,10,11,12,13,14,15);
    }
#pragma unroll
    for (int j = 0; j < 4; ++j) {
      const bf16* p = &shB[bufi][(wn + j * 16 + l16) * 40 + kb];
      bf16x8 lo = *(const bf16x8*)p;
      bf16x8 hi = *(const bf16x8*)(p + 16);
      bfr[j] = __builtin_shufflevector(lo, hi, 0,1,2,3,4,5,6,7,8,9,10,11,12,13,14,15);
    }
#pragma unroll
    for (int i = 0; i < 2; ++i)
#pragma unroll
      for (int j = 0; j < 4; ++j)
        acc[i][j] = __builtin_amdgcn_wmma_f32_16x16x32_bf16(
            false, af[i], false, bfr[j], (short)0, acc[i][j], false, false);
  };

  const int KT = K >> 5;
  issue_tiles(0, 0);
  int buf = 0;
  for (int kt = 0; kt < KT - 1; ++kt) {
    wait_async0();              // this wave's tile-kt LDS writes done (in-order)
    __syncthreads();            // all waves' writes visible; all readers of buf^1 done
    issue_tiles(buf ^ 1, (kt + 1) * 32);   // fill next buffer during compute
    compute(buf);
    buf ^= 1;
  }
  wait_async0();
  __syncthreads();
  compute(buf);

  // Epilogue. C layout: lane<16 -> M = base+r, lane>=16 -> M = base+8+r; N = base+l16.
#pragma unroll
  for (int i = 0; i < 2; ++i) {
    int rbase = row0 + wm + i * 16 + half * 8;
#pragma unroll
    for (int j = 0; j < 4; ++j) {
      int col = col0 + wn + j * 16 + l16;
      float bv = HAS_BIAS ? bias[col] : 0.0f;
#pragma unroll
      for (int r = 0; r < 8; ++r) {
        float v = acc[i][j][r] * scale + bv;
        if (RELU) v = v > 0.0f ? v : 0.0f;
        size_t idx = (size_t)(rbase + r) * ldc + col;
        if (OUT_BF16) Cb[idx] = f2bf(v);
        else          Cf[idx] = v;
      }
    }
  }
}

// ---------------------------------------------------------------------------
// fp32 -> bf16 elementwise cast (4 elems / thread)
// ---------------------------------------------------------------------------
__global__ __launch_bounds__(256)
void cast_bf16(const float* __restrict__ in, bf16* __restrict__ out, size_t n)
{
  size_t i = ((size_t)blockIdx.x * 256 + threadIdx.x) * 4;
  if (i + 3 < n) {
#pragma unroll
    for (int t = 0; t < 4; ++t) out[i + t] = f2bf(in[i + t]);
  }
}

// ---------------------------------------------------------------------------
// fp32 [rows x cols] -> bf16 transposed [cols x rows]   (weights: Wt[n][k]=W[k][n])
// block (32,8), grid (cols/32, rows/32)
// ---------------------------------------------------------------------------
__global__ __launch_bounds__(256)
void transpose_cast_bf16(const float* __restrict__ in, bf16* __restrict__ out,
                         int rows, int cols)
{
  __shared__ float tile[32][33];
  int tx = threadIdx.x, ty = threadIdx.y;
  int c0 = blockIdx.x * 32, r0 = blockIdx.y * 32;
#pragma unroll
  for (int i = 0; i < 4; ++i)
    tile[ty + i * 8][tx] = in[(size_t)(r0 + ty + i * 8) * cols + c0 + tx];
  __syncthreads();
#pragma unroll
  for (int i = 0; i < 4; ++i)
    out[(size_t)(c0 + ty + i * 8) * rows + r0 + tx] = f2bf(tile[tx][ty + i * 8]);
}

// ---------------------------------------------------------------------------
// bf16 [rows x cols] -> bf16 transposed [cols x rows], batched (for v -> v^T)
// ---------------------------------------------------------------------------
__global__ __launch_bounds__(256)
void transpose_bf16(const bf16* __restrict__ in, bf16* __restrict__ out,
                    int rows, int cols, long long bsIn, long long bsOut)
{
  in  += (long long)blockIdx.z * bsIn;
  out += (long long)blockIdx.z * bsOut;
  __shared__ bf16 tile[32][34];
  int tx = threadIdx.x, ty = threadIdx.y;
  int c0 = blockIdx.x * 32, r0 = blockIdx.y * 32;
#pragma unroll
  for (int i = 0; i < 4; ++i)
    tile[ty + i * 8][tx] = in[(size_t)(r0 + ty + i * 8) * cols + c0 + tx];
  __syncthreads();
#pragma unroll
  for (int i = 0; i < 4; ++i)
    out[(size_t)(c0 + ty + i * 8) * rows + r0 + tx] = tile[tx][ty + i * 8];
}

// ---------------------------------------------------------------------------
// Row softmax over T=2048 f32 scores -> bf16 probabilities. 1 block / row.
// ---------------------------------------------------------------------------
__global__ __launch_bounds__(256)
void softmax_bf16(const float* __restrict__ S, bf16* __restrict__ P)
{
  __shared__ float red[256];
  size_t row = blockIdx.x;
  const float* src = S + row * 2048;
  bf16* dst = P + row * 2048;
  int tid = threadIdx.x;

  float x[8];
  float mx = -3.0e38f;
#pragma unroll
  for (int i = 0; i < 8; ++i) { x[i] = src[tid + i * 256]; mx = fmaxf(mx, x[i]); }
  red[tid] = mx; __syncthreads();
  for (int s = 128; s > 0; s >>= 1) {
    if (tid < s) red[tid] = fmaxf(red[tid], red[tid + s]);
    __syncthreads();
  }
  mx = red[0]; __syncthreads();

  float sum = 0.0f;
#pragma unroll
  for (int i = 0; i < 8; ++i) {
    x[i] = exp2f((x[i] - mx) * 1.4426950408889634f);
    sum += x[i];
  }
  red[tid] = sum; __syncthreads();
  for (int s = 128; s > 0; s >>= 1) {
    if (tid < s) red[tid] += red[tid + s];
    __syncthreads();
  }
  float inv = 1.0f / red[0];
#pragma unroll
  for (int i = 0; i < 8; ++i) dst[tid + i * 256] = f2bf(x[i] * inv);
}

// ---------------------------------------------------------------------------
// Fused residual + LayerNorm over D=1024. out (f32) and optional bf16 copy.
// ---------------------------------------------------------------------------
template<bool WB>
__global__ __launch_bounds__(256)
void layernorm_res(const float* __restrict__ Y, const float* __restrict__ R,
                   const float* __restrict__ g, const float* __restrict__ b,
                   float* __restrict__ out, bf16* __restrict__ outb)
{
  __shared__ float red[256];
  size_t row = blockIdx.x;
  const float* y = Y + row * 1024;
  const float* r = R + row * 1024;
  int tid = threadIdx.x;

  float x[4];
  float s = 0.0f;
#pragma unroll
  for (int i = 0; i < 4; ++i) { x[i] = y[tid + i * 256] + r[tid + i * 256]; s += x[i]; }
  red[tid] = s; __syncthreads();
  for (int st = 128; st > 0; st >>= 1) {
    if (tid < st) red[tid] += red[tid + st];
    __syncthreads();
  }
  float mu = red[0] * (1.0f / 1024.0f); __syncthreads();

  float v = 0.0f;
#pragma unroll
  for (int i = 0; i < 4; ++i) { float d = x[i] - mu; v += d * d; }
  red[tid] = v; __syncthreads();
  for (int st = 128; st > 0; st >>= 1) {
    if (tid < st) red[tid] += red[tid + st];
    __syncthreads();
  }
  float rstd = rsqrtf(red[0] * (1.0f / 1024.0f) + 1e-5f);

#pragma unroll
  for (int i = 0; i < 4; ++i) {
    int c = tid + i * 256;
    float o = (x[i] - mu) * rstd * g[c] + b[c];
    out[row * 1024 + c] = o;
    if (WB) outb[row * 1024 + c] = f2bf(o);
  }
}

// ---------------------------------------------------------------------------
// Host-side orchestration
// ---------------------------------------------------------------------------
extern "C" void kernel_launch(void* const* d_in, const int* in_sizes, int n_in,
                              void* d_out, int out_size, void* d_ws, size_t ws_size,
                              hipStream_t stream) {
  constexpr int N = 4, T = 2048, D = 1024;
  constexpr long long NT  = (long long)N * T;          // 8192
  constexpr long long NTD = NT * D;                    // 8.4M
  constexpr long long DDl = (long long)D * D;          // 1M
  constexpr long long NTT = (long long)N * T * T;      // 16.8M

  const float* X  = (const float*)d_in[0];
  const float* Wq = (const float*)d_in[1];  const float* bq = (const float*)d_in[2];
  const float* Wk = (const float*)d_in[3];  const float* bk = (const float*)d_in[4];
  const float* Wv = (const float*)d_in[5];  const float* bv = (const float*)d_in[6];
  const float* W1 = (const float*)d_in[7];  const float* b1 = (const float*)d_in[8];
  const float* W2 = (const float*)d_in[9];  const float* b2 = (const float*)d_in[10];
  const float* W3 = (const float*)d_in[11]; const float* b3 = (const float*)d_in[12];
  const float* g1 = (const float*)d_in[13]; const float* be1 = (const float*)d_in[14];
  const float* g2 = (const float*)d_in[15]; const float* be2 = (const float*)d_in[16];
  float* out = (float*)d_out;

  char* ws = (char*)d_ws;
  size_t off = 0;
  auto alloc = [&](size_t bytes) -> void* {
    void* p = ws + off;
    off = (off + bytes + 255) & ~(size_t)255;
    return p;
  };

  bf16* Xbf = (bf16*)alloc(NTD * 2);
  bf16* WqT = (bf16*)alloc(DDl * 2);
  bf16* WkT = (bf16*)alloc(DDl * 2);
  bf16* WvT = (bf16*)alloc(DDl * 2);
  bf16* W1T = (bf16*)alloc(DDl * 2);
  bf16* W2T = (bf16*)alloc(DDl * 2);
  bf16* W3T = (bf16*)alloc(DDl * 2);
  bf16* qb  = (bf16*)alloc(NTD * 2);
  bf16* kb_ = (bf16*)alloc(NTD * 2);
  bf16* vb  = (bf16*)alloc(NTD * 2);
  bf16* vT  = (bf16*)alloc(NTD * 2);
  float* sc = (float*)alloc(NTT * 4);
  bf16* at  = (bf16*)alloc(NTT * 2);
  float* cx = (float*)alloc(NTD * 4);    // ctx, reused for m3
  float* hf = (float*)alloc(NTD * 4);
  bf16* hb  = (bf16*)alloc(NTD * 2);
  bf16* m1  = (bf16*)alloc(NTD * 2);
  bf16* m2  = qb;                        // q is dead after scores GEMM
  float* m3 = cx;                        // ctx is dead after LN1

  dim3 blk256(256);
  dim3 blkT(32, 8);

  // Stage bf16 operands (weights pre-transposed: Wt[n][k] = W[k][n]).
  cast_bf16<<<dim3((unsigned)(NTD / 1024)), blk256, 0, stream>>>(X, Xbf, (size_t)NTD);
  dim3 gW(D / 32, D / 32);
  transpose_cast_bf16<<<gW, blkT, 0, stream>>>(Wq, WqT, D, D);
  transpose_cast_bf16<<<gW, blkT, 0, stream>>>(Wk, WkT, D, D);
  transpose_cast_bf16<<<gW, blkT, 0, stream>>>(Wv, WvT, D, D);
  transpose_cast_bf16<<<gW, blkT, 0, stream>>>(W1, W1T, D, D);
  transpose_cast_bf16<<<gW, blkT, 0, stream>>>(W2, W2T, D, D);
  transpose_cast_bf16<<<gW, blkT, 0, stream>>>(W3, W3T, D, D);

  // QKV projections: [8192x1024] = Xbf @ W, fused bias, bf16 out.
  dim3 gProj(D / 128, (unsigned)(NT / 128), 1);
  gemm_wmma<true, false, true><<<gProj, blk256, 0, stream>>>(
      Xbf, WqT, bq, nullptr, qb, (int)NT, D, D, D, D, D, 0, 0, 0, 1.0f);
  gemm_wmma<true, false, true><<<gProj, blk256, 0, stream>>>(
      Xbf, WkT, bk, nullptr, kb_, (int)NT, D, D, D, D, D, 0, 0, 0, 1.0f);
  gemm_wmma<true, false, true><<<gProj, blk256, 0, stream>>>(
      Xbf, WvT, bv, nullptr, vb, (int)NT, D, D, D, D, D, 0, 0, 0, 1.0f);

  // v^T per batch for the PV GEMM (B operand must be [Ncol x K]).
  dim3 gVT(D / 32, T / 32, N);
  transpose_bf16<<<gVT, blkT, 0, stream>>>(vb, vT, T, D, (long long)T * D, (long long)D * T);

  // scores = q @ k^T / sqrt(D), batched, f32 out.  (k rows are already [s][d] = Bt)
  dim3 gS(T / 128, T / 128, N);
  gemm_wmma<false, false, false><<<gS, blk256, 0, stream>>>(
      qb, kb_, nullptr, sc, nullptr, T, T, D, D, D, T,
      (long long)T * D, (long long)T * D, (long long)T * T, 0.03125f);

  // row softmax -> bf16 attn
  softmax_bf16<<<dim3((unsigned)(N * T)), blk256, 0, stream>>>(sc, at);

  // ctx = attn @ v  (Bt = v^T), batched, f32 out.
  dim3 gC(D / 128, T / 128, N);
  gemm_wmma<false, false, false><<<gC, blk256, 0, stream>>>(
      at, vT, nullptr, cx, nullptr, T, D, T, T, T, D,
      (long long)T * T, (long long)D * T, (long long)T * D, 1.0f);

  // h = LN(ctx + X); emit f32 (for residual 2) + bf16 (FFN input)
  layernorm_res<true><<<dim3((unsigned)NT), blk256, 0, stream>>>(cx, X, g1, be1, hf, hb);

  // FFN
  gemm_wmma<true, true, true><<<gProj, blk256, 0, stream>>>(
      hb, W1T, b1, nullptr, m1, (int)NT, D, D, D, D, D, 0, 0, 0, 1.0f);
  gemm_wmma<true, true, true><<<gProj, blk256, 0, stream>>>(
      m1, W2T, b2, nullptr, m2, (int)NT, D, D, D, D, D, 0, 0, 0, 1.0f);
  gemm_wmma<false, false, true><<<gProj, blk256, 0, stream>>>(
      m2, W3T, b3, m3, nullptr, (int)NT, D, D, D, D, D, 0, 0, 0, 1.0f);

  // out = LN(m3 + h)
  layernorm_res<false><<<dim3((unsigned)NT), blk256, 0, stream>>>(m3, hf, g2, be2, out, nullptr);

  (void)in_sizes; (void)n_in; (void)out_size; (void)ws_size;
}
